// QuantumLSTMGNN_70592082477727
// MI455X (gfx1250) — matrix operand order repaired and checked by reading, not compiled, not run on hardware
//
#include <hip/hip_runtime.h>

// ---------------------------------------------------------------------------
// QuantumLSTMGNN for MI455X (gfx1250, wave32, WMMA)
// Pipeline: GATv2 x2 -> BiLSTM -> Linear+ELU -> classifier
// Dense GEMMs use v_wmma_f32_16x16x32_bf16 (bf16 in, f32 accum) with
// pre-transposed bf16 weights so both A and B fragments are contiguous
// 2x b128 loads per lane, and an explicit 2-stage software pipeline so
// loads for the next k-chunk overlap the current chunk's WMMAs.
// The serial LSTM scan double-buffers per-step gate rows in LDS via
// GLOBAL_LOAD_ASYNC_TO_LDS_B128 (ASYNCcnt + s_wait_asynccnt).
// ---------------------------------------------------------------------------

#define NN   4096
#define EE   65536
#define IND  14
#define HD   256
#define H1C  4
#define H2C  2

typedef __attribute__((ext_vector_type(16))) __bf16 v16bf;
typedef __attribute__((ext_vector_type(8)))  float  v8f;

__device__ __forceinline__ unsigned short f2bf(float f) {
    unsigned u = __float_as_uint(f);
    unsigned r = u + 0x7FFFu + ((u >> 16) & 1u);   // round-to-nearest-even
    return (unsigned short)(r >> 16);
}
__device__ __forceinline__ float sigm(float x) { return 1.f / (1.f + __expf(-x)); }

__device__ __forceinline__ v16bf frag(uint4 x, uint4 y) {
    union { uint4 q[2]; v16bf v; } u; u.q[0] = x; u.q[1] = y; return u.v;
}

__device__ __forceinline__ void atomicMaxF(float* addr, float v) {
    unsigned int* ua = reinterpret_cast<unsigned int*>(addr);
    unsigned int cur = __float_as_uint(*addr);
    while (__uint_as_float(cur) < v) {
        unsigned int prev = atomicCAS(ua, cur, __float_as_uint(v));
        if (prev == cur) break;
        cur = prev;
    }
}

// --- CDNA5 async global->LDS copy (per-lane b128), tracked by ASYNCcnt -----
__device__ __forceinline__ void async_load_b128(void* lds_dst, const void* gsrc) {
    // flat LDS aperture pointer: addr[31:0] is the in-allocation byte offset
    unsigned ldsoff = (unsigned)(unsigned long long)(uintptr_t)lds_dst;
    unsigned long long ga = (unsigned long long)(uintptr_t)gsrc;
    asm volatile("global_load_async_to_lds_b128 %0, %1, off"
                 :: "v"(ldsoff), "v"(ga) : "memory");
}
__device__ __forceinline__ void wait_async0() {
    asm volatile("s_wait_asynccnt 0x0" ::: "memory");
}

// ---------------------------------------------------------------------------
// Utility kernels
// ---------------------------------------------------------------------------
__global__ void fillf(float* __restrict__ p, float v, long long n) {
    long long i = (long long)blockIdx.x * blockDim.x + threadIdx.x;
    if (i < n) p[i] = v;
}

// weight convert + transpose: Bt[n,k] = bf16(W[k,n]);  W is [K,Nn]
__global__ void cvt_bf16_t(const float* __restrict__ W, unsigned short* __restrict__ Bt,
                           int K, int Nn) {
    long long idx = (long long)blockIdx.x * blockDim.x + threadIdx.x;
    if (idx >= (long long)K * Nn) return;
    int n = (int)(idx / K), k = (int)(idx % K);
    Bt[idx] = f2bf(W[(size_t)k * Nn + n]);
}

// Y[n,j] = b[j] + sum_k X[n,k] * W[k,j]   (tiny K, e.g. K=14)
__global__ void lin_small(const float* __restrict__ X, const float* __restrict__ W,
                          const float* __restrict__ b, float* __restrict__ Y,
                          int rows, int K, int Nn) {
    long long idx = (long long)blockIdx.x * blockDim.x + threadIdx.x;
    if (idx >= (long long)rows * Nn) return;
    int n = (int)(idx / Nn), j = (int)(idx % Nn);
    float s = b[j];
    const float* xr = X + (size_t)n * K;
    #pragma unroll 7
    for (int k = 0; k < K; ++k) s += xr[k] * W[(size_t)k * Nn + j];
    Y[idx] = s;
}

// ---------------------------------------------------------------------------
// WMMA bf16 GEMM: C[M,Nn] = act(A[M,K] @ Bt[Nn,K]^T + bias[Nn])
// A row-major [M,K]; Bt pre-transposed to [Nn,K] so each lane's fragment is
// 32 contiguous bytes (2x global_load_b128), same as A.
// Wave tile 16(M)x64(N), 8 waves/block arranged 2x4 -> block tile 32x256.
// Explicit 2-stage pipeline over 64-wide k-chunks: fragments of chunk i+1
// are loaded into distinct registers while chunk i's WMMAs execute.
// Requires M%32==0, Nn%256==0, K%64==0 (true for all call sites).
// act: 0=none, 1=relu, 2=elu
// Fragment layouts used here (A/B: contiguous k order, same for both; not the ISA table order):
//   A (16x32 bf16): lane m=lane&15, hi=lane>=16; halves 0..15 = K=hi*16+0..15
//   B (32x16 bf16): lane n=lane&15; halves 0..15 = K=hi*16+0..15
//   C/D (16x16 f32): vgpr r, lane l: row = r + 8*(l>=16), col = l&15
// ---------------------------------------------------------------------------
__global__ __launch_bounds__(256) void gemm_bf16_wmma(
    const unsigned short* __restrict__ A,
    const unsigned short* __restrict__ Bt,
    const float* __restrict__ bias,
    float* __restrict__ C,
    int M, int Nn, int K, int act)
{
    const int tid  = threadIdx.x;
    const int wave = tid >> 5;
    const int lane = tid & 31;
    const int wm   = wave & 1;
    const int wn   = wave >> 1;
    const int m0   = blockIdx.y * 32 + wm * 16;
    const int n0   = blockIdx.x * 256 + wn * 64;
    const int hi   = lane >> 4;       // half-wave select
    const int lr   = lane & 15;

    v8f acc[4];
    #pragma unroll
    for (int t = 0; t < 4; ++t)
        acc[t] = (v8f){0.f, 0.f, 0.f, 0.f, 0.f, 0.f, 0.f, 0.f};

    const unsigned short* Ab = A  + (size_t)(m0 + lr) * K + (hi << 4);
    const unsigned short* Bb = Bt + (size_t)(n0 + lr) * K + (hi << 4);
    const size_t bstr = (size_t)16 * K;     // Bt stride per 16-wide n-tile

    uint4 aq[2][2];       // [stage][half]
    uint4 bq[2][4][2];    // [stage][n-tile][half]

    // prologue: stage 0 <- k=0
    aq[0][0] = *(const uint4*)(Ab);
    aq[0][1] = *(const uint4*)(Ab + 8);
    #pragma unroll
    for (int t = 0; t < 4; ++t) {
        const unsigned short* p = Bb + t * bstr;
        bq[0][t][0] = *(const uint4*)(p);
        bq[0][t][1] = *(const uint4*)(p + 8);
    }

    for (int k0 = 0; k0 < K; k0 += 64) {
        // stage 1 <- k0+32 (in flight during stage-0 WMMAs)
        {
            const unsigned short* pa = Ab + k0 + 32;
            aq[1][0] = *(const uint4*)(pa);
            aq[1][1] = *(const uint4*)(pa + 8);
            #pragma unroll
            for (int t = 0; t < 4; ++t) {
                const unsigned short* p = Bb + t * bstr + k0 + 32;
                bq[1][t][0] = *(const uint4*)(p);
                bq[1][t][1] = *(const uint4*)(p + 8);
            }
        }
        // compute stage 0
        #pragma unroll
        for (int t = 0; t < 4; ++t)
            acc[t] = __builtin_amdgcn_wmma_f32_16x16x32_bf16(
                false, frag(aq[0][0], aq[0][1]), false, frag(bq[0][t][0], bq[0][t][1]),
                (short)0, acc[t], false, false);
        // stage 0 <- k0+64 (in flight during stage-1 WMMAs); uniform guard
        if (k0 + 64 < K) {
            const unsigned short* pa = Ab + k0 + 64;
            aq[0][0] = *(const uint4*)(pa);
            aq[0][1] = *(const uint4*)(pa + 8);
            #pragma unroll
            for (int t = 0; t < 4; ++t) {
                const unsigned short* p = Bb + t * bstr + k0 + 64;
                bq[0][t][0] = *(const uint4*)(p);
                bq[0][t][1] = *(const uint4*)(p + 8);
            }
        }
        // compute stage 1
        #pragma unroll
        for (int t = 0; t < 4; ++t)
            acc[t] = __builtin_amdgcn_wmma_f32_16x16x32_bf16(
                false, frag(aq[1][0], aq[1][1]), false, frag(bq[1][t][0], bq[1][t][1]),
                (short)0, acc[t], false, false);
    }

    // store with bias + activation
    #pragma unroll
    for (int t = 0; t < 4; ++t) {
        const int col = n0 + t * 16 + lr;
        const float bv = bias[col];
        #pragma unroll
        for (int r = 0; r < 8; ++r) {
            const int row = m0 + (hi << 3) + r;
            float v = acc[t][r] + bv;
            if (act == 1)      v = fmaxf(v, 0.f);
            else if (act == 2) v = (v > 0.f) ? v : (__expf(v) - 1.f);
            C[(size_t)row * Nn + col] = v;
        }
    }
}

// ---------------------------------------------------------------------------
// GATv2 edge kernels
// ---------------------------------------------------------------------------
__global__ __launch_bounds__(256) void edge_logits(
    const float* __restrict__ xl, const float* __restrict__ xr,
    const float* __restrict__ att,
    const int* __restrict__ src, const int* __restrict__ dst,
    float* __restrict__ logits, float* __restrict__ segmax,
    int H, int HC)
{
    const int gw   = (blockIdx.x * 256 + threadIdx.x) >> 5;
    const int lane = threadIdx.x & 31;
    const int e = gw / H, h = gw % H;
    const int s = src[e], d = dst[e];
    const float* pl = xl + (size_t)s * HC + h * HD;
    const float* pr = xr + (size_t)d * HC + h * HD;
    const float* pa = att + h * HD;
    float sum = 0.f;
    #pragma unroll
    for (int i = lane; i < HD; i += 32) {
        float v = pl[i] + pr[i];
        v = (v > 0.f) ? v : 0.2f * v;          // leaky_relu, slope 0.2
        sum += v * pa[i];
    }
    #pragma unroll
    for (int off = 16; off > 0; off >>= 1) sum += __shfl_xor(sum, off, 32);
    if (lane == 0) {
        logits[gw] = sum;
        atomicMaxF(&segmax[d * H + h], sum);
    }
}

__global__ void edge_expsum(const float* __restrict__ logits,
                            const float* __restrict__ segmax,
                            const int* __restrict__ dst,
                            float* __restrict__ pbuf, float* __restrict__ segsum,
                            int H) {
    int idx = blockIdx.x * blockDim.x + threadIdx.x;
    if (idx >= EE * H) return;
    int e = idx / H, h = idx % H;
    int d = dst[e];
    float p = __expf(logits[idx] - segmax[d * H + h]);
    pbuf[idx] = p;
    atomicAdd(&segsum[d * H + h], p);
}

__global__ void edge_scatter(const float* __restrict__ xl,
                             const float* __restrict__ pbuf,
                             const float* __restrict__ segsum,
                             const int* __restrict__ src, const int* __restrict__ dst,
                             float* __restrict__ agg, int H, int HC) {
    long long idx = (long long)blockIdx.x * blockDim.x + threadIdx.x;
    if (idx >= (long long)EE * HC) return;
    int e = (int)(idx / HC), ch = (int)(idx % HC);
    int h = ch >> 8;                            // HD = 256
    int s = src[e], d = dst[e];
    float alpha = pbuf[e * H + h] / (segsum[d * H + h] + 1e-16f);
    atomicAdd(&agg[(size_t)d * HC + ch], xl[(size_t)s * HC + ch] * alpha);
}

__global__ void bias_relu_bf16(const float* __restrict__ agg, const float* __restrict__ bias,
                               unsigned short* __restrict__ out, long long n, int Nn) {
    long long i = (long long)blockIdx.x * blockDim.x + threadIdx.x;
    if (i >= n) return;
    float v = agg[i] + bias[(int)(i % Nn)];
    out[i] = f2bf(fmaxf(v, 0.f));
}

// ---------------------------------------------------------------------------
// Bidirectional LSTM scan: 2 blocks (fwd/bwd), 1024 threads each.
// gates_in precomputed by WMMA GEMM. Per-step 4KB gate row is double-buffered
// in LDS via global_load_async_to_lds_b128 (ASYNCcnt), overlapping the next
// step's fetch with the current recurrent matvec (the serial critical path).
// ---------------------------------------------------------------------------
__global__ __launch_bounds__(1024) void lstm_scan(
    const float* __restrict__ gates_f, const float* __restrict__ gates_b,
    const float* __restrict__ Whh_f,   const float* __restrict__ Whh_b,
    unsigned short* __restrict__ hcatb, int T)
{
    const int dir = blockIdx.x;
    const float* gates = dir ? gates_b : gates_f;
    const float* Whh   = dir ? Whh_b   : Whh_f;
    const int j = threadIdx.x;
    __shared__ float gbuf[2][4 * HD];
    __shared__ float h_sh[HD];
    __shared__ float g_sh[4 * HD];
    float c = 0.f;
    if (j < HD) h_sh[j] = 0.f;

    const int t0 = dir ? (T - 1) : 0;
    if (j < 256)  // waves 0..7, wave-uniform predicate
        async_load_b128(&gbuf[0][j * 4], gates + (size_t)t0 * (4 * HD) + j * 4);
    wait_async0();
    __syncthreads();

    int pb = 0;
    for (int s = 0; s < T; ++s) {
        const int t = dir ? (T - 1 - s) : s;
        if (s + 1 < T) {
            const int tn = dir ? (T - 2 - s) : (s + 1);
            if (j < 256)
                async_load_b128(&gbuf[pb ^ 1][j * 4], gates + (size_t)tn * (4 * HD) + j * 4);
        }
        float g = gbuf[pb][j];
        #pragma unroll 8
        for (int k = 0; k < HD; ++k) g += h_sh[k] * Whh[(size_t)k * (4 * HD) + j];
        g_sh[j] = g;
        __syncthreads();
        if (j < HD) {
            float ig = g_sh[j], fg = g_sh[HD + j], gg = g_sh[2 * HD + j], og = g_sh[3 * HD + j];
            c = sigm(fg) * c + sigm(ig) * tanhf(gg);
            float hn = sigm(og) * tanhf(c);
            h_sh[j] = hn;
            hcatb[(size_t)t * (2 * HD) + dir * HD + j] = f2bf(hn);
        }
        wait_async0();        // next-step gates resident before buffer swap
        __syncthreads();
        pb ^= 1;
    }
}

// ---------------------------------------------------------------------------
// Classifier: out[n,j<5] = bc[j] + sum_k q[n,k]*Wc[k,j]
// ---------------------------------------------------------------------------
__global__ void classifier(const float* __restrict__ q, const float* __restrict__ Wc,
                           const float* __restrict__ bc, float* __restrict__ out) {
    int idx = blockIdx.x * blockDim.x + threadIdx.x;
    if (idx >= NN * 5) return;
    int n = idx / 5, j = idx % 5;
    float s = bc[j];
    const float* qr = q + (size_t)n * 1024;
    #pragma unroll 8
    for (int k = 0; k < 1024; ++k) s += qr[k] * Wc[k * 5 + j];
    out[idx] = s;
}

// ---------------------------------------------------------------------------
// Host launch
// ---------------------------------------------------------------------------
extern "C" void kernel_launch(void* const* d_in, const int* in_sizes, int n_in,
                              void* d_out, int out_size, void* d_ws, size_t ws_size,
                              hipStream_t stream) {
    const float* x     = (const float*)d_in[0];
    const int*   eidx  = (const int*)  d_in[1];
    const float* Wl1   = (const float*)d_in[2];
    const float* bl1   = (const float*)d_in[3];
    const float* Wr1   = (const float*)d_in[4];
    const float* br1   = (const float*)d_in[5];
    const float* att1  = (const float*)d_in[6];
    const float* bo1   = (const float*)d_in[7];
    const float* Wl2   = (const float*)d_in[8];
    const float* bl2   = (const float*)d_in[9];
    const float* Wr2   = (const float*)d_in[10];
    const float* br2   = (const float*)d_in[11];
    const float* att2  = (const float*)d_in[12];
    const float* bo2   = (const float*)d_in[13];
    const float* Wih_f = (const float*)d_in[14];
    const float* Whh_f = (const float*)d_in[15];
    const float* b_f   = (const float*)d_in[16];
    const float* Wih_b = (const float*)d_in[17];
    const float* Whh_b = (const float*)d_in[18];
    const float* b_b   = (const float*)d_in[19];
    const float* Wq    = (const float*)d_in[20];
    const float* bq    = (const float*)d_in[21];
    const float* Wc    = (const float*)d_in[22];
    const float* bc    = (const float*)d_in[23];
    float* out = (float*)d_out;

    const int* src = eidx;
    const int* dst = eidx + EE;

    char* ws = (char*)d_ws;
    size_t off = 0;
    auto alloc = [&](size_t bytes) -> void* {
        void* p = ws + off;
        off = (off + bytes + 255) & ~(size_t)255;
        return p;
    };
    float* xl1   = (float*)alloc((size_t)NN * 1024 * 4);
    float* xr1   = (float*)alloc((size_t)NN * 1024 * 4);
    float* agg1  = (float*)alloc((size_t)NN * 1024 * 4);
    unsigned short* h1b = (unsigned short*)alloc((size_t)NN * 1024 * 2);
    float* xl2   = (float*)alloc((size_t)NN * 512 * 4);
    float* xr2   = (float*)alloc((size_t)NN * 512 * 4);
    float* agg2  = (float*)alloc((size_t)NN * 512 * 4);
    unsigned short* h2b = (unsigned short*)alloc((size_t)NN * 512 * 2);
    float* gatesF = (float*)alloc((size_t)NN * 1024 * 4);
    float* gatesB = (float*)alloc((size_t)NN * 1024 * 4);
    unsigned short* hcatb = (unsigned short*)alloc((size_t)NN * 512 * 2);
    float* q     = (float*)alloc((size_t)NN * 1024 * 4);
    float* logitsB = (float*)alloc((size_t)EE * H1C * 4);
    float* pbuf    = (float*)alloc((size_t)EE * H1C * 4);
    float* segmax  = (float*)alloc((size_t)NN * H1C * 4);
    float* segsum  = (float*)alloc((size_t)NN * H1C * 4);
    unsigned short* Wl2b  = (unsigned short*)alloc((size_t)1024 * 512 * 2);  // [512,1024] transposed
    unsigned short* Wr2b  = (unsigned short*)alloc((size_t)1024 * 512 * 2);
    unsigned short* Wihfb = (unsigned short*)alloc((size_t)512 * 1024 * 2);  // [1024,512] transposed
    unsigned short* Wihbb = (unsigned short*)alloc((size_t)512 * 1024 * 2);
    unsigned short* Wqb   = (unsigned short*)alloc((size_t)512 * 1024 * 2);

    const int TB = 256;
    auto blk = [](long long n, int t) { return (unsigned)((n + t - 1) / t); };
    const float NEG_INF = -__builtin_inff();

    // --- weight convert + transpose (bf16, [Nn,K] layout for WMMA B) ---
    cvt_bf16_t<<<blk(1024 * 512, TB), TB, 0, stream>>>(Wl2,   Wl2b,  1024, 512);
    cvt_bf16_t<<<blk(1024 * 512, TB), TB, 0, stream>>>(Wr2,   Wr2b,  1024, 512);
    cvt_bf16_t<<<blk(512 * 1024, TB), TB, 0, stream>>>(Wih_f, Wihfb, 512, 1024);
    cvt_bf16_t<<<blk(512 * 1024, TB), TB, 0, stream>>>(Wih_b, Wihbb, 512, 1024);
    cvt_bf16_t<<<blk(512 * 1024, TB), TB, 0, stream>>>(Wq,    Wqb,   512, 1024);

    // --- GATv2 layer 1 (K=14 transform: scalar kernel; WMMA would waste K) ---
    lin_small<<<blk((long long)NN * 1024, TB), TB, 0, stream>>>(x, Wl1, bl1, xl1, NN, IND, 1024);
    lin_small<<<blk((long long)NN * 1024, TB), TB, 0, stream>>>(x, Wr1, br1, xr1, NN, IND, 1024);
    fillf<<<blk(NN * H1C, TB), TB, 0, stream>>>(segmax, NEG_INF, NN * H1C);
    fillf<<<blk(NN * H1C, TB), TB, 0, stream>>>(segsum, 0.f, NN * H1C);
    fillf<<<blk((long long)NN * 1024, TB), TB, 0, stream>>>(agg1, 0.f, (long long)NN * 1024);
    edge_logits<<<(EE * H1C) / 8, TB, 0, stream>>>(xl1, xr1, att1, src, dst, logitsB, segmax, H1C, 1024);
    edge_expsum<<<blk(EE * H1C, TB), TB, 0, stream>>>(logitsB, segmax, dst, pbuf, segsum, H1C);
    edge_scatter<<<blk((long long)EE * 1024, TB), TB, 0, stream>>>(xl1, pbuf, segsum, src, dst, agg1, H1C, 1024);
    bias_relu_bf16<<<blk((long long)NN * 1024, TB), TB, 0, stream>>>(agg1, bo1, h1b, (long long)NN * 1024, 1024);

    // --- GATv2 layer 2 transforms (WMMA): [4096,1024]@[1024,512] ---
    gemm_bf16_wmma<<<dim3(512 / 256, NN / 32), 256, 0, stream>>>(h1b, Wl2b, bl2, xl2, NN, 512, 1024, 0);
    gemm_bf16_wmma<<<dim3(512 / 256, NN / 32), 256, 0, stream>>>(h1b, Wr2b, br2, xr2, NN, 512, 1024, 0);
    fillf<<<blk(NN * H2C, TB), TB, 0, stream>>>(segmax, NEG_INF, NN * H2C);
    fillf<<<blk(NN * H2C, TB), TB, 0, stream>>>(segsum, 0.f, NN * H2C);
    fillf<<<blk((long long)NN * 512, TB), TB, 0, stream>>>(agg2, 0.f, (long long)NN * 512);
    edge_logits<<<(EE * H2C) / 8, TB, 0, stream>>>(xl2, xr2, att2, src, dst, logitsB, segmax, H2C, 512);
    edge_expsum<<<blk(EE * H2C, TB), TB, 0, stream>>>(logitsB, segmax, dst, pbuf, segsum, H2C);
    edge_scatter<<<blk((long long)EE * 512, TB), TB, 0, stream>>>(xl2, pbuf, segsum, src, dst, agg2, H2C, 512);
    bias_relu_bf16<<<blk((long long)NN * 512, TB), TB, 0, stream>>>(agg2, bo2, h2b, (long long)NN * 512, 512);

    // --- LSTM input projections (WMMA): [4096,512]@[512,1024] ---
    gemm_bf16_wmma<<<dim3(1024 / 256, NN / 32), 256, 0, stream>>>(h2b, Wihfb, b_f, gatesF, NN, 1024, 512, 0);
    gemm_bf16_wmma<<<dim3(1024 / 256, NN / 32), 256, 0, stream>>>(h2b, Wihbb, b_b, gatesB, NN, 1024, 512, 0);

    // --- sequential BiLSTM scan (fwd + bwd concurrently, async gate prefetch) ---
    lstm_scan<<<2, 1024, 0, stream>>>(gatesF, gatesB, Whh_f, Whh_b, hcatb, NN);

    // --- quantum layer (WMMA + ELU): [4096,512]@[512,1024] ---
    gemm_bf16_wmma<<<dim3(1024 / 256, NN / 32), 256, 0, stream>>>(hcatb, Wqb, bq, q, NN, 1024, 512, 2);

    // --- classifier ---
    classifier<<<blk(NN * 5, TB), TB, 0, stream>>>(q, Wc, bc, out);
}
